// GatingNetwork_53798760350314
// MI455X (gfx1250) — matrix operand order repaired and checked by reading, not compiled, not run on hardware
//
#include <hip/hip_runtime.h>
#include <hip/hip_bf16.h>
#include <math.h>

// MI455X / gfx1250 gating network.
// Roofline: (16384x2048)@(2048x64) = 4.3 GFLOP over 128MB fp32 reads ->
// ~5.5us HBM floor @ 23.3 TB/s. bf16 WMMA (16x16x32, f32 acc) keeps the GEMM
// above the bandwidth-implied 780 TFLOPS requirement; fp32 WMMA would not.
// This version removes all intra-loop barriers/LDS: each wave owns 16 rows x
// all 64 experts, loads its A fragment straight from global in a contiguous-K (k = 16h + s)
// layout, so the compiler can pipeline loads across iterations.

typedef __bf16 bf16_t;
typedef __bf16 v16bf __attribute__((ext_vector_type(16)));
typedef float  v8f   __attribute__((ext_vector_type(8)));
typedef float  v4f   __attribute__((ext_vector_type(4)));

#define NEXP 64
#define WPB  4                       // waves per block
#define RPW  16                      // rows per wave (WMMA M)
#define RPB  (WPB * RPW)             // 64 rows per block
#define LPAD 68                      // LDS row stride (floats): conflict-free

// ---------------------------------------------------------------------------
// Prep: column-normalize sim_matrix, fold experts_mask into the bf16 B matrix
// (column-major, 256KB, L2-resident), and emit gates*sigmoid(temp) thresholds.
// ---------------------------------------------------------------------------
__global__ void __launch_bounds__(256)
gating_prep(const float* __restrict__ sim,
            const float* __restrict__ gates,
            const float* __restrict__ temperature,
            const float* __restrict__ emask,
            bf16_t* __restrict__ Bws,
            float* __restrict__ gthresh,
            int C, int E) {
  __shared__ float red[256];
  const int j = blockIdx.x;
  const int t = threadIdx.x;

  float ss = 0.0f;
  for (int k = t; k < C; k += 256) {
    float v = sim[(size_t)k * E + j];
    ss = fmaf(v, v, ss);
  }
  red[t] = ss;
  __syncthreads();
  for (int off = 128; off > 0; off >>= 1) {
    if (t < off) red[t] += red[t + off];
    __syncthreads();
  }
  const float scale = (1.0f / fmaxf(sqrtf(red[0]), 1e-12f)) * emask[j];
  for (int k = t; k < C; k += 256) {
    Bws[(size_t)j * C + k] = (bf16_t)(sim[(size_t)k * E + j] * scale);
  }
  if (t == 0) {
    const float logit_scale = 1.0f / (1.0f + __expf(-temperature[0]));
    gthresh[j] = gates[j] * logit_scale;
  }
}

// ---------------------------------------------------------------------------
// Main: 4 independent waves per block; wave owns rows [base, base+16) and all
// 64 experts (4 accumulators). No barriers in the K loop.
// ---------------------------------------------------------------------------
__global__ void __launch_bounds__(WPB * 32)
gating_main(const float* __restrict__ X,
            const bf16_t* __restrict__ Bmat,
            const float* __restrict__ gthresh,
            const int* __restrict__ kminp,
            float* __restrict__ outMask,
            float* __restrict__ outLogits,
            int C) {
  __shared__ float Llog[WPB][RPW][LPAD];   // per-wave logits slab (~17KB)

  const int t    = threadIdx.x;
  const int lane = t & 31;
  const int w    = t >> 5;
  const int h    = lane >> 4;      // lane half: K-interleave select
  const int m    = lane & 15;      // row within wave tile / N within 16-tile

  const long long rowBase = (long long)blockIdx.x * RPB + (long long)w * RPW;

  // A: lane half h owns K = [16h,16h+16) (e0..15)
  // of row (rowBase+m). Lanes m and m+16 cover disjoint K -> A read once.
  const float* xLo = X + (size_t)(rowBase + m) * C + 16 * h;

  // B: lane = column (m), 16 contiguous K per lane half; 4 expert tiles.
  const bf16_t* bBase = Bmat + (size_t)m * C + 16 * h;
  const size_t  bTile = (size_t)16 * C;    // column stride between et tiles

  v8f acc0 = {0,0,0,0,0,0,0,0};
  v8f acc1 = {0,0,0,0,0,0,0,0};
  v8f acc2 = {0,0,0,0,0,0,0,0};
  v8f acc3 = {0,0,0,0,0,0,0,0};
  float ss = 0.0f;

  const int nk = C >> 5;                   // K-steps of 32
#pragma unroll 2
  for (int kt = 0; kt < nk; ++kt) {
    const int k0 = kt << 5;

    // ---- A fragment straight from global (fp32), fused row sum-of-squares
    v4f a0 = *(const v4f*)(xLo + k0);          // K = k0+16h   .. +3
    v4f a1 = *(const v4f*)(xLo + k0 + 4);      // K = k0+16h+4 .. +7
    v4f a2 = *(const v4f*)(xLo + k0 + 8);      // K = k0+16h+8 ..
    v4f a3 = *(const v4f*)(xLo + k0 + 12);     // K = k0+16h+12 ..
    ss = fmaf(a0.x,a0.x, fmaf(a0.y,a0.y, fmaf(a0.z,a0.z, fmaf(a0.w,a0.w, ss))));
    ss = fmaf(a1.x,a1.x, fmaf(a1.y,a1.y, fmaf(a1.z,a1.z, fmaf(a1.w,a1.w, ss))));
    ss = fmaf(a2.x,a2.x, fmaf(a2.y,a2.y, fmaf(a2.z,a2.z, fmaf(a2.w,a2.w, ss))));
    ss = fmaf(a3.x,a3.x, fmaf(a3.y,a3.y, fmaf(a3.z,a3.z, fmaf(a3.w,a3.w, ss))));

    v16bf a = {(bf16_t)a0.x,(bf16_t)a0.y,(bf16_t)a0.z,(bf16_t)a0.w,
               (bf16_t)a1.x,(bf16_t)a1.y,(bf16_t)a1.z,(bf16_t)a1.w,
               (bf16_t)a2.x,(bf16_t)a2.y,(bf16_t)a2.z,(bf16_t)a2.w,
               (bf16_t)a3.x,(bf16_t)a3.y,(bf16_t)a3.z,(bf16_t)a3.w};

    // ---- B fragments (L2-resident), one per 16-expert tile
    v16bf b0 = *(const v16bf*)(bBase + k0);
    v16bf b1 = *(const v16bf*)(bBase + bTile     + k0);
    v16bf b2 = *(const v16bf*)(bBase + bTile * 2 + k0);
    v16bf b3 = *(const v16bf*)(bBase + bTile * 3 + k0);

    acc0 = __builtin_amdgcn_wmma_f32_16x16x32_bf16(false, a, false, b0, (short)0, acc0, false, false);
    acc1 = __builtin_amdgcn_wmma_f32_16x16x32_bf16(false, a, false, b1, (short)0, acc1, false, false);
    acc2 = __builtin_amdgcn_wmma_f32_16x16x32_bf16(false, a, false, b2, (short)0, acc2, false, false);
    acc3 = __builtin_amdgcn_wmma_f32_16x16x32_bf16(false, a, false, b3, (short)0, acc3, false, false);
  }

  // ---- row inverse norms: lanes m and m+16 hold the two K-halves of row m
  ss += __shfl_xor(ss, 16, 32);
  const float rinv = 1.0f / fmaxf(sqrtf(ss), 1e-12f);   // F.normalize eps

  // ---- scale (C/D layout: VGPR v -> row v+8h) and stage to LDS
#pragma unroll
  for (int v = 0; v < 8; ++v) {
    const float sc = __shfl(rinv, v + 8 * h, 32);  // lane r holds row r's rinv
    const int r = 8 * h + v;
    Llog[w][r][ 0 + m] = acc0[v] * sc;
    Llog[w][r][16 + m] = acc1[v] * sc;
    Llog[w][r][32 + m] = acc2[v] * sc;
    Llog[w][r][48 + m] = acc3[v] * sc;
  }
  __syncthreads();

  // ---- per-row epilogue: lanes 0..15 of each wave own one row each
  if (lane < 16) {
    const long long r = rowBase + lane;
    float* gl = outLogits + r * NEXP;
#pragma unroll
    for (int i = 0; i < 16; ++i) {         // conflict-free: bank = 4*lane+...
      *(v4f*)(gl + 4 * i) = *(const v4f*)&Llog[w][lane][4 * i];
    }

    unsigned long long bits = 0ull;
    int cnt = 0;
#pragma unroll
    for (int j = 0; j < NEXP; ++j) {
      const float v = Llog[w][lane][j];
      if (v > gthresh[j]) { bits |= 1ull << j; ++cnt; }  // relu(l-g)>0 <=> l>g
    }
    if (cnt == 0) {
      int kmin = *kminp;
      if (kmin > NEXP) kmin = NEXP;
      unsigned long long chosen = 0ull;
      for (int it = 0; it < kmin; ++it) {
        float bv = -INFINITY;
        int bi = -1;
#pragma unroll
        for (int j = 0; j < NEXP; ++j) {
          if ((chosen >> j) & 1ull) continue;
          const float v = Llog[w][lane][j];
          if (v > bv) { bv = v; bi = j; }  // strict >: ties -> lowest index
        }
        if (bi >= 0) chosen |= 1ull << bi;
      }
      bits = chosen;
    }

    float* gm = outMask + r * NEXP;
#pragma unroll
    for (int i = 0; i < 16; ++i) {
      v4f mv = {(float)((bits >> (4 * i    )) & 1ull),
                (float)((bits >> (4 * i + 1)) & 1ull),
                (float)((bits >> (4 * i + 2)) & 1ull),
                (float)((bits >> (4 * i + 3)) & 1ull)};
      *(v4f*)(gm + 4 * i) = mv;
    }
  }
}

// ---------------------------------------------------------------------------
extern "C" void kernel_launch(void* const* d_in, const int* in_sizes, int n_in,
                              void* d_out, int out_size, void* d_ws, size_t ws_size,
                              hipStream_t stream) {
  const float* X     = (const float*)d_in[0];   // hidden_states (B,T,C) f32
  const float* sim   = (const float*)d_in[1];   // sim_matrix (C,E) f32
  const float* gates = (const float*)d_in[2];   // gates (E,)
  const float* temp  = (const float*)d_in[3];   // temperature (1,)
  const float* emask = (const float*)d_in[4];   // experts_mask (E,)
  const int*   kmin  = (const int*)d_in[5];     // min_experts_per_tok

  const int E = in_sizes[2];                    // 64
  const int C = in_sizes[1] / E;                // 2048
  const long long N = (long long)in_sizes[0] / C;  // 16384 rows

  bf16_t* Bws     = (bf16_t*)d_ws;                                  // C*E bf16
  float*  gthresh = (float*)((char*)d_ws + (size_t)C * E * sizeof(bf16_t));

  float* outMask   = (float*)d_out;
  float* outLogits = outMask + (size_t)N * E;

  gating_prep<<<E, 256, 0, stream>>>(sim, gates, temp, emask, Bws, gthresh, C, E);

  gating_main<<<(int)(N / RPB), WPB * 32, 0, stream>>>(
      X, Bws, gthresh, kmin, outMask, outLogits, C);
}
